// ACmix_58798102282697
// MI455X (gfx1250) — matrix operand order repaired and checked by reading, not compiled, not run on hardware
//
#include <hip/hip_runtime.h>
#include <hip/hip_bf16.h>

typedef __bf16 bf16;
typedef __attribute__((ext_vector_type(16))) __bf16 v16bf;
typedef __attribute__((ext_vector_type(8)))  __bf16 v8bf;
typedef __attribute__((ext_vector_type(8)))  float  v8f;

// ---------------------------------------------------------------------------
// Fragment helpers. LDS tiles stored [row][k], k contiguous, row stride 40
// bf16 (80B, 16B-aligned) so each WMMA fragment is two 16-byte LDS loads
// matching the gfx1250 16-bit A/B VGPR layouts:
//   A (16x32, MxK): lane -> row = lane&15, khalf = (lane>>4)*8
//       elems 0..7 -> K = khalf+0..7 ; elems 8..15 -> K = 16+khalf+0..7
//   B (32x16, KxN): lane -> col = lane&15, khalf = (lane>>4)*8
//       elems 0..7 -> K = khalf+0..7 ; elems 8..15 -> K = 16+khalf+0..7
// ---------------------------------------------------------------------------
__device__ __forceinline__ v16bf make_frag(const bf16* lo, const bf16* hi) {
  v8bf l = *(const v8bf*)lo;
  v8bf h = *(const v8bf*)hi;
  v16bf r;
#pragma unroll
  for (int i = 0; i < 8; ++i) { r[i] = l[i]; r[i + 8] = h[i]; }
  return r;
}

__device__ __forceinline__ v8f wmma_bf16(v16bf a, v16bf b, v8f c) {
  return __builtin_amdgcn_wmma_f32_16x16x32_bf16(false, a, false, b,
                                                 (short)0, c, false, false);
}

__device__ __forceinline__ v8f zero_v8f() {
  v8f z = {0.f, 0.f, 0.f, 0.f, 0.f, 0.f, 0.f, 0.f};
  return z;
}

// ---------------------------------------------------------------------------
// pos[c,h,w] = convp_w[c%64,0]*lx(w) + convp_w[c%64,1]*ly(h) + convp_b[c%64]
// ---------------------------------------------------------------------------
__global__ __launch_bounds__(256) void pos_kernel(
    const float* __restrict__ pw, const float* __restrict__ pb,
    float* __restrict__ pos) {
  int idx = blockIdx.x * 256 + threadIdx.x;      // 256*64*64 = 1,048,576
  if (idx >= 256 * 4096) return;
  int c = idx >> 12, hw = idx & 4095, h = hw >> 6, w = hw & 63;
  int cc = c & 63;
  float lx = -1.0f + (2.0f / 63.0f) * (float)w;
  float ly = -1.0f + (2.0f / 63.0f) * (float)h;
  pos[idx] = pw[cc * 2 + 0] * lx + pw[cc * 2 + 1] * ly + pb[cc];
}

// ---------------------------------------------------------------------------
// 256x256 channel-mix GEMM: Y[m,n] = sum_c W[m,c]*X[c,n] + fused epilogue.
// X layout [batch][256][S], S = 1<<ls. mode 0: +bias  1: relu(y*s+b)  2: raw
// 128x128 block tile, double-buffered bf16 LDS, one barrier per K-chunk.
// Grid (Ntotal/128, 2), 256 threads = 8 waves in a 4(M)x2(N) grid.
// ---------------------------------------------------------------------------
__global__ __launch_bounds__(256) void gemm256_wmma(
    const float* __restrict__ X, const float* __restrict__ Wm,
    const float* __restrict__ p0, const float* __restrict__ p1,
    float* __restrict__ Y, int ls, int mode) {
  __shared__ bf16 As[2][128][40];
  __shared__ bf16 Bs[2][128][40];

  const int tid = threadIdx.x, lane = tid & 31, wid = tid >> 5;
  const int n0 = blockIdx.x * 128, m0 = blockIdx.y * 128;
  const int wm = wid & 3, wn = wid >> 2;
  const int Smask = (1 << ls) - 1;

  v8f acc[2][4];
#pragma unroll
  for (int im = 0; im < 2; ++im)
#pragma unroll
    for (int in = 0; in < 4; ++in) acc[im][in] = zero_v8f();

  const int arow = tid >> 1, akoff = (tid & 1) * 16;   // A: 2 thr/row, 16 k
  const int bk = tid >> 3, bnoff = (tid & 7) * 16;     // B: 8 thr/k-row, 16 n

  // Hoisted per-thread global base pointers; each K-chunk strides by 32 rows.
  const float* wp = Wm + (size_t)(m0 + arow) * 256 + akoff;
  const int nb = n0 + bnoff;
  const int nbatch = nb >> ls, nsc = nb & Smask;
  const float* xp = X + (((size_t)nbatch * 256 + bk) << ls) + nsc;
  const size_t xstride = (size_t)32 << ls;             // 32 channels / chunk

  auto stage = [&](int pp, int kc) {
    const float* w = wp + kc * 32;
    const float* xq = xp + (size_t)kc * xstride;
#pragma unroll
    for (int i = 0; i < 16; i += 4) {
      float4 wv = *(const float4*)(w + i);             // global_load_b128
      As[pp][arow][akoff + i + 0] = (bf16)wv.x;
      As[pp][arow][akoff + i + 1] = (bf16)wv.y;
      As[pp][arow][akoff + i + 2] = (bf16)wv.z;
      As[pp][arow][akoff + i + 3] = (bf16)wv.w;
    }
#pragma unroll
    for (int i = 0; i < 16; i += 4) {                  // 16-aligned n-run:
      float4 xv = *(const float4*)(xq + i);            // never crosses batch
      Bs[pp][bnoff + i + 0][bk] = (bf16)xv.x;
      Bs[pp][bnoff + i + 1][bk] = (bf16)xv.y;
      Bs[pp][bnoff + i + 2][bk] = (bf16)xv.z;
      Bs[pp][bnoff + i + 3][bk] = (bf16)xv.w;
    }
  };

  stage(0, 0);
  __syncthreads();

  for (int kc = 0; kc < 8; ++kc) {
    const int pp = kc & 1;
    if (kc < 7) stage(pp ^ 1, kc + 1);                 // overlap with wmma
    if (kc < 6) __builtin_prefetch(xp + (size_t)(kc + 2) * xstride, 0, 0);

    const int khalf = (lane >> 4) << 3;
    const int k16   = (lane >> 4) << 4;
    const int r     = lane & 15;
    v16bf afr[2], bfr[4];
#pragma unroll
    for (int im = 0; im < 2; ++im) {
      int row = wm * 32 + im * 16 + r;
      afr[im] = make_frag(&As[pp][row][khalf], &As[pp][row][16 + khalf]);
    }
#pragma unroll
    for (int in = 0; in < 4; ++in) {
      int col = wn * 64 + in * 16 + r;
      bfr[in] = make_frag(&Bs[pp][col][khalf], &Bs[pp][col][16 + khalf]);
    }
#pragma unroll
    for (int im = 0; im < 2; ++im)
#pragma unroll
      for (int in = 0; in < 4; ++in)
        acc[im][in] = wmma_bf16(afr[im], bfr[in], acc[im][in]);
    __syncthreads();                                   // one barrier / chunk
  }

  // Fused epilogue; base address hoisted per tile, row stride = S floats.
#pragma unroll
  for (int im = 0; im < 2; ++im)
#pragma unroll
    for (int in = 0; in < 4; ++in) {
      const int mbase = m0 + wm * 32 + im * 16 + ((lane >> 4) << 3);
      const int n = n0 + wn * 64 + in * 16 + (lane & 15);
      const int bt = n >> ls, sc = n & Smask;
      float* yp = Y + (((size_t)bt * 256 + mbase) << ls) + sc;
#pragma unroll
      for (int j = 0; j < 8; ++j) {
        float v = acc[im][in][j];
        const int m = mbase + j;
        if (mode == 0)      v += p0[m];
        else if (mode == 1) v = fmaxf(v * p0[m] + p1[m], 0.0f);
        yp[(size_t)j << ls] = v;
      }
    }
}

// ---------------------------------------------------------------------------
// Batched self-attention, one block per batch item (1024 blocks).
//   S = softmax((Q^T K) * C^-0.5),  O = V S^T   (C=256, N=M=64)
// mode 1: long-range — fuses blockify permutation + q*hd^-0.5+pos / k+pos
// gathers; mode 0: contiguous [bb][256][64]. Output contiguous.
// LDS arena (47KB): Q/K double-buffered chunks overlaid later by V chunks.
// ---------------------------------------------------------------------------
__global__ __launch_bounds__(256) void sa_wmma(
    const float* __restrict__ Qg, const float* __restrict__ Kg,
    const float* __restrict__ Vg, const float* __restrict__ Pg,
    float* __restrict__ Og, int mode) {
  __shared__ __align__(16) unsigned char smem[47104];
  typedef bf16 row40[40];
  row40* QK = (row40*)smem;                         // [2][64] Q + [2][64] K
  row40* Vc = (row40*)smem;                         // stage-2 overlay [256]
  float (*Sf)[68] = (float(*)[68])(smem + 20480);   // 17408B
  bf16 (*Pb)[72]  = (bf16(*)[72])(smem + 37888);    // 9216B

  const int bb = blockIdx.x;
  const int tid = threadIdx.x, lane = tid & 31, wid = tid >> 5;

  int gb = 0, dh = 0, dw = 0;
  if (mode) { gb = bb >> 6; int rr = bb & 63; dh = rr >> 3; dw = rr & 7; }

  // ---- Stage 1: scores S[n][m] = sum_c q[c][n]*k[c][m], K chunked by 32 ----
  const int tn_w = wid & 3, tm2 = (wid >> 2) * 2;
  const int ln = tid >> 2, coff = (tid & 3) * 8;    // loader: row ln, 8 c's
  int hw_n = 0;
  if (mode) hw_n = ((ln >> 3) * 8 + dh) * 64 + ((ln & 7) * 8 + dw);

  auto stage1 = [&](int pp, int kc) {
    const int c0 = kc * 32 + coff;
    bf16* qrow = QK[pp * 64 + ln] + coff;
    bf16* krow = QK[128 + pp * 64 + ln] + coff;
    if (mode == 0) {
      const float* qp = Qg + ((size_t)bb * 256 + c0) * 64 + ln;
      const float* kp = Kg + ((size_t)bb * 256 + c0) * 64 + ln;
#pragma unroll
      for (int i = 0; i < 8; ++i) {
        qrow[i] = (bf16)qp[(size_t)i * 64];
        krow[i] = (bf16)kp[(size_t)i * 64];
      }
    } else {
      const float* qp = Qg + ((size_t)gb * 256 + c0) * 4096 + hw_n;
      const float* kp = Kg + ((size_t)gb * 256 + c0) * 4096 + hw_n;
      const float* pp2 = Pg + (size_t)c0 * 4096 + hw_n;
#pragma unroll
      for (int i = 0; i < 8; ++i) {
        float pv = pp2[(size_t)i * 4096];
        qrow[i] = (bf16)(qp[(size_t)i * 4096] * 0.125f + pv);
        krow[i] = (bf16)(kp[(size_t)i * 4096] + pv);
      }
    }
  };

  v8f sacc[2] = {zero_v8f(), zero_v8f()};
  stage1(0, 0);
  __syncthreads();
  for (int kc = 0; kc < 8; ++kc) {
    const int pp = kc & 1;
    if (kc < 7) stage1(pp ^ 1, kc + 1);
    const int khalf = (lane >> 4) << 3, k16 = (lane >> 4) << 4, r = lane & 15;
    const bf16* qr = QK[pp * 64 + tn_w * 16 + r];
    v16bf aq = make_frag(qr + khalf, qr + 16 + khalf);
#pragma unroll
    for (int t = 0; t < 2; ++t) {
      const bf16* kr = QK[128 + pp * 64 + (tm2 + t) * 16 + r];
      v16bf bk = make_frag(kr + khalf, kr + 16 + khalf);
      sacc[t] = wmma_bf16(aq, bk, sacc[t]);
    }
    __syncthreads();
  }
#pragma unroll
  for (int t = 0; t < 2; ++t)
#pragma unroll
    for (int j = 0; j < 8; ++j) {
      int n = tn_w * 16 + j + ((lane >> 4) << 3);
      int m = (tm2 + t) * 16 + (lane & 15);
      Sf[n][m] = sacc[t][j] * 0.0625f;              // * C^-0.5 = 1/16
    }
  __syncthreads();

  // ---- Softmax over m, rows n = wid*8..wid*8+7, wave32 reductions ----
#pragma unroll
  for (int rr = 0; rr < 8; ++rr) {
    int n = wid * 8 + rr;
    float a = Sf[n][lane], b2 = Sf[n][lane + 32];
    float mx = fmaxf(a, b2);
#pragma unroll
    for (int o = 16; o > 0; o >>= 1) mx = fmaxf(mx, __shfl_xor(mx, o, 32));
    float e0 = __expf(a - mx), e1 = __expf(b2 - mx);
    float sm = e0 + e1;
#pragma unroll
    for (int o = 16; o > 0; o >>= 1) sm += __shfl_xor(sm, o, 32);
    float inv = 1.0f / sm;
    Pb[n][lane]      = (bf16)(e0 * inv);
    Pb[n][lane + 32] = (bf16)(e1 * inv);
  }
  __syncthreads();   // Q/K region now dead -> safe to overlay Vc

  // ---- Stage 2: O[c][n] = sum_m V[c][m]*P[n][m], K = 64 in 2 chunks ----
  v8f oacc[2][4];
#pragma unroll
  for (int t = 0; t < 2; ++t)
#pragma unroll
    for (int in = 0; in < 4; ++in) oacc[t][in] = zero_v8f();

  for (int mc = 0; mc < 2; ++mc) {
    const int mb = mc * 32;
    if (mode == 0) {
      const float4* vp = (const float4*)(Vg + ((size_t)bb * 256 + tid) * 64 + mb);
#pragma unroll
      for (int i4 = 0; i4 < 8; ++i4) {
        float4 v4 = vp[i4];
        Vc[tid][i4 * 4 + 0] = (bf16)v4.x;
        Vc[tid][i4 * 4 + 1] = (bf16)v4.y;
        Vc[tid][i4 * 4 + 2] = (bf16)v4.z;
        Vc[tid][i4 * 4 + 3] = (bf16)v4.w;
      }
    } else {
      const float* vp = Vg + ((size_t)gb * 256 + tid) * 4096 + dh * 64 + dw;
#pragma unroll
      for (int i = 0; i < 32; ++i) {
        int m = mb + i;
        Vc[tid][i] = (bf16)vp[(m >> 3) * 512 + (m & 7) * 8];
      }
    }
    __syncthreads();
    const int khalf = (lane >> 4) << 3, k16 = (lane >> 4) << 4, r = lane & 15;
    v16bf av[2];
#pragma unroll
    for (int t = 0; t < 2; ++t) {
      const bf16* vr = Vc[(wid * 2 + t) * 16 + r];
      av[t] = make_frag(vr + khalf, vr + 16 + khalf);
    }
#pragma unroll
    for (int in = 0; in < 4; ++in) {
      const bf16* pr = Pb[in * 16 + r];
      v16bf bp = make_frag(pr + mb + khalf, pr + mb + 16 + khalf);
#pragma unroll
      for (int t = 0; t < 2; ++t) oacc[t][in] = wmma_bf16(av[t], bp, oacc[t][in]);
    }
    __syncthreads();
  }
#pragma unroll
  for (int t = 0; t < 2; ++t)
#pragma unroll
    for (int in = 0; in < 4; ++in) {
      const int cb2 = (wid * 2 + t) * 16 + ((lane >> 4) << 3);
      const int n = in * 16 + (lane & 15);
      float* op = Og + ((size_t)bb * 256 + cb2) * 64 + n;
#pragma unroll
      for (int j = 0; j < 8; ++j) op[(size_t)j * 64] = oacc[t][in][j];
    }
}

// ---------------------------------------------------------------------------
// feats regroup: [b*64(dh,dw)][C][8x8(i,j)] -> [b*64(i,j)][C][8x8(dh,dw)]
// ---------------------------------------------------------------------------
__global__ __launch_bounds__(256) void regroup_kernel(
    const float* __restrict__ src, float* __restrict__ dst) {
  int idx = blockIdx.x * 256 + threadIdx.x;      // 16,777,216
  int s2 = idx & 63;
  int c = (idx >> 6) & 255;
  int bb = idx >> 14;
  int b = bb >> 6, ij = bb & 63, i = ij >> 3, j = ij & 7;
  int dh = s2 >> 3, dwv = s2 & 7;
  size_t s = ((size_t)(b * 64 + dh * 8 + dwv) * 256 + c) * 64 + (i * 8 + j);
  dst[idx] = src[s];
}

// ---------------------------------------------------------------------------
// Conv branch fc: f_conv[b][d*9+o][hw] = sum_i cat(q,k,v)[b][i][d][hw]*fc_w[o,i]
// ---------------------------------------------------------------------------
__global__ __launch_bounds__(256) void fc_kernel(
    const float* __restrict__ Q, const float* __restrict__ K,
    const float* __restrict__ V, const float* __restrict__ fcw,
    float* __restrict__ F) {
  int idx = blockIdx.x * 256 + threadIdx.x;      // 16*64*4096 = 4,194,304
  int hw = idx & 4095;
  int d = (idx >> 12) & 63;
  int b = idx >> 18;
  float in[12];
#pragma unroll
  for (int i = 0; i < 4; ++i) {
    size_t base = ((size_t)b * 256 + i * 64 + d) * 4096 + hw;
    in[i] = Q[base]; in[4 + i] = K[base]; in[8 + i] = V[base];
  }
#pragma unroll
  for (int o = 0; o < 9; ++o) {
    float s = 0.f;
#pragma unroll
    for (int i = 0; i < 12; ++i) s += in[i] * fcw[o * 12 + i];
    F[((size_t)b * 576 + d * 9 + o) * 4096 + hw] = s;
  }
}

// ---------------------------------------------------------------------------
// Grouped 3x3 conv (64 groups, 9 in / 4 out per group), pad 1.
// Writes d_out = rate2 * out_conv (first writer of d_out).
// ---------------------------------------------------------------------------
__global__ __launch_bounds__(256) void dwconv_kernel(
    const float* __restrict__ F, const float* __restrict__ dwt,
    const float* __restrict__ rate2, float* __restrict__ out) {
  int idx = blockIdx.x * 256 + threadIdx.x;      // 16*256*4096
  int w = idx & 63, h = (idx >> 6) & 63;
  int oc = (idx >> 12) & 255;
  int b = idx >> 20;
  int g = oc >> 2;
  float s = 0.f;
#pragma unroll
  for (int p = 0; p < 9; ++p) {
    const float* fp = F + ((size_t)b * 576 + g * 9 + p) * 4096;
#pragma unroll
    for (int kh = 0; kh < 3; ++kh) {
      int hh = h + kh - 1;
      if (hh < 0 || hh > 63) continue;
#pragma unroll
      for (int kw = 0; kw < 3; ++kw) {
        int ww = w + kw - 1;
        if (ww < 0 || ww > 63) continue;
        s += fp[hh * 64 + ww] * dwt[((oc * 9 + p) * 3 + kh) * 3 + kw];
      }
    }
  }
  out[idx] = rate2[0] * s;
}

// ---------------------------------------------------------------------------
// d_out[b][c][h][w] += rate1 * feats2[(b*64+i*8+j)][c][dh*8+dw]
// ---------------------------------------------------------------------------
__global__ __launch_bounds__(256) void addatt_kernel(
    const float* __restrict__ feats2, const float* __restrict__ rate1,
    float* __restrict__ out) {
  int idx = blockIdx.x * 256 + threadIdx.x;      // 16*256*4096
  int w = idx & 63, h = (idx >> 6) & 63;
  int c = (idx >> 12) & 255;
  int b = idx >> 20;
  int i = h >> 3, dh = h & 7, j = w >> 3, dwv = w & 7;
  size_t src = ((size_t)(b * 64 + i * 8 + j) * 256 + c) * 64 + (dh * 8 + dwv);
  out[idx] += rate1[0] * feats2[src];
}

// ---------------------------------------------------------------------------
extern "C" void kernel_launch(void* const* d_in, const int* in_sizes, int n_in,
                              void* d_out, int out_size, void* d_ws,
                              size_t ws_size, hipStream_t stream) {
  (void)in_sizes; (void)n_in; (void)out_size; (void)ws_size;
  const float* x       = (const float*)d_in[0];
  const float* conv1_w = (const float*)d_in[1];
  const float* conv1_b = (const float*)d_in[2];
  const float* conv2_w = (const float*)d_in[3];
  const float* conv2_b = (const float*)d_in[4];
  const float* conv3_w = (const float*)d_in[5];
  const float* conv3_b = (const float*)d_in[6];
  const float* convp_w = (const float*)d_in[7];
  const float* convp_b = (const float*)d_in[8];
  const float* fc_w    = (const float*)d_in[9];
  const float* dep_w   = (const float*)d_in[10];
  const float* rate1   = (const float*)d_in[11];
  const float* rate2   = (const float*)d_in[12];
  const float* lrW_w   = (const float*)d_in[13];
  const float* lrW_s   = (const float*)d_in[14];
  const float* lrW_sh  = (const float*)d_in[15];
  const float* fq1_w   = (const float*)d_in[16];
  const float* fq1_s   = (const float*)d_in[17];
  const float* fq1_sh  = (const float*)d_in[18];
  const float* fq2_w   = (const float*)d_in[19];
  const float* fq2_s   = (const float*)d_in[20];
  const float* fq2_sh  = (const float*)d_in[21];
  const float* fk1_w   = (const float*)d_in[22];
  const float* fk1_s   = (const float*)d_in[23];
  const float* fk1_sh  = (const float*)d_in[24];
  const float* fk2_w   = (const float*)d_in[25];
  const float* fk2_s   = (const float*)d_in[26];
  const float* fk2_sh  = (const float*)d_in[27];
  const float* fv_w    = (const float*)d_in[28];
  const float* srW_w   = (const float*)d_in[29];
  const float* srW_s   = (const float*)d_in[30];
  const float* srW_sh  = (const float*)d_in[31];

  float* out = (float*)d_out;
  float* ws  = (float*)d_ws;

  const size_t P = 1048576;          // 256*64*64
  const size_t T = 16777216;         // 1024*256*64 == 16*256*64*64
  float* pos = ws;
  float* Qb  = ws + P;
  float* Kb  = Qb + T;
  float* Vb  = Kb + T;
  float* A   = Vb + T;
  float* B   = A + T;
  float* Cb  = B + T;
  float* Db  = Cb + T;
  float* F   = A;                    // 151MB f_conv overlays A..C (dead later)

  const dim3 gg(512, 2);             // 65536/128 N-tiles, 2 M-tiles

  // positional encoding
  pos_kernel<<<4096, 256, 0, stream>>>(convp_w, convp_b, pos);

  // q, k, v = 1x1 convs (WMMA GEMMs, fused bias)
  gemm256_wmma<<<gg, 256, 0, stream>>>(x, conv1_w, conv1_b, nullptr, Qb, 12, 0);
  gemm256_wmma<<<gg, 256, 0, stream>>>(x, conv2_w, conv2_b, nullptr, Kb, 12, 0);
  gemm256_wmma<<<gg, 256, 0, stream>>>(x, conv3_w, conv3_b, nullptr, Vb, 12, 0);

  // conv branch first (its big f_conv buffer overlays attention scratch)
  fc_kernel<<<16384, 256, 0, stream>>>(Qb, Kb, Vb, fc_w, F);
  dwconv_kernel<<<65536, 256, 0, stream>>>(F, dep_w, rate2, out);

  // attention branch: long-range SA (fused blockify + pos + hd^-0.5 gather)
  sa_wmma<<<1024, 256, 0, stream>>>(Qb, Kb, Vb, pos, A, 1);
  gemm256_wmma<<<gg, 256, 0, stream>>>(A, lrW_w, lrW_s, lrW_sh, B, 6, 1);
  regroup_kernel<<<65536, 256, 0, stream>>>(B, Cb);

  // short-range q/k/v projections (fused BN+ReLU epilogues)
  gemm256_wmma<<<gg, 256, 0, stream>>>(Cb, fq1_w, fq1_s, fq1_sh, A, 6, 1);
  gemm256_wmma<<<gg, 256, 0, stream>>>(A, fq2_w, fq2_s, fq2_sh, B, 6, 1);   // qx
  gemm256_wmma<<<gg, 256, 0, stream>>>(Cb, fk1_w, fk1_s, fk1_sh, A, 6, 1);
  gemm256_wmma<<<gg, 256, 0, stream>>>(A, fk2_w, fk2_s, fk2_sh, Db, 6, 1);  // kx
  gemm256_wmma<<<gg, 256, 0, stream>>>(Cb, fv_w, nullptr, nullptr, A, 6, 2);// vx

  // short-range SA + output projection, then combine with rate1
  sa_wmma<<<1024, 256, 0, stream>>>(B, Db, A, nullptr, Cb, 0);
  gemm256_wmma<<<gg, 256, 0, stream>>>(Cb, srW_w, srW_s, srW_sh, A, 6, 1);
  addatt_kernel<<<65536, 256, 0, stream>>>(A, rate1, out);
}
